// SelectiveScanModule_3410204033448
// MI455X (gfx1250) — matrix operand order repaired and checked by reading, not compiled, not run on hardware
//
#include <hip/hip_runtime.h>

// ---------------------------------------------------------------------------
// Selective scan (Mamba-style) for MI455X / gfx1250, wave32 + WMMA bf16.
// B=8, S=4096, Dm=1024, N=16.  Dominant op: y @ W_o^T (68.7 GFLOP) -> WMMA,
// with async global->LDS double-buffered B tiles (ASYNCcnt path).
// ---------------------------------------------------------------------------

#define BB   8
#define SS   4096
#define DM   1024
#define NN   16
#define ROWS (BB * SS)          // 32768

typedef __bf16 bf16x16 __attribute__((ext_vector_type(16)));
typedef __bf16 bf16x8  __attribute__((ext_vector_type(8)));
typedef float  v8f     __attribute__((ext_vector_type(8)));

__device__ __forceinline__ unsigned short f2bf(float f) {
    unsigned u = __float_as_uint(f);
    unsigned r = u + 0x7FFFu + ((u >> 16) & 1u);   // round-to-nearest-even
    return (unsigned short)(r >> 16);
}

// ---------------------------------------------------------------------------
// Kernel 1: delta, Bu, A_exp.  One wave per (b,s) row.
// ---------------------------------------------------------------------------
__global__ void k_bu_delta(const float* __restrict__ x,
                           const float* __restrict__ A,
                           const float* __restrict__ Wb,
                           const float* __restrict__ bb,
                           const float* __restrict__ Wt,
                           const float* __restrict__ bt,
                           float* __restrict__ Bu,
                           float* __restrict__ Aexp) {
    int wave = blockIdx.x * (blockDim.x >> 5) + (threadIdx.x >> 5);
    int lane = threadIdx.x & 31;
    if (wave >= ROWS) return;

    const float* xr = x + (size_t)wave * DM;

    float at = 0.0f;
    float acc[NN];
#pragma unroll
    for (int n = 0; n < NN; ++n) acc[n] = 0.0f;

    for (int j = 0; j < DM / 32; ++j) {
        int d = lane + j * 32;
        float xv = xr[d];
        at = fmaf(xv, Wt[d], at);
#pragma unroll
        for (int n = 0; n < NN; ++n)
            acc[n] = fmaf(xv, Wb[n * DM + d], acc[n]);
    }

#pragma unroll
    for (int off = 16; off >= 1; off >>= 1) {
        at += __shfl_xor(at, off, 32);
#pragma unroll
        for (int n = 0; n < NN; ++n)
            acc[n] += __shfl_xor(acc[n], off, 32);
    }

    float z     = at + bt[0];
    float sig   = 1.0f / (1.0f + __expf(-z));
    float delta = sig * 0.99f + 0.01f;

    if (lane == 0) {
        float4* bu4 = (float4*)(Bu   + (size_t)wave * NN);
        float4* ae4 = (float4*)(Aexp + (size_t)wave * NN);
#pragma unroll
        for (int q = 0; q < 4; ++q) {
            float4 bv, av;
            bv.x = acc[4 * q + 0] + bb[4 * q + 0];
            bv.y = acc[4 * q + 1] + bb[4 * q + 1];
            bv.z = acc[4 * q + 2] + bb[4 * q + 2];
            bv.w = acc[4 * q + 3] + bb[4 * q + 3];
            av.x = __expf(A[4 * q + 0] * delta);
            av.y = __expf(A[4 * q + 1] * delta);
            av.z = __expf(A[4 * q + 2] * delta);
            av.w = __expf(A[4 * q + 3] * delta);
            bu4[q] = bv;
            ae4[q] = av;
        }
    }
}

// ---------------------------------------------------------------------------
// Kernel 2: associative scan  h_t = Aexp_t * h_{t-1} + Bu_t.
// ---------------------------------------------------------------------------
__global__ void k_scan(const float* __restrict__ Aexp,
                       const float* __restrict__ Bu,
                       float* __restrict__ H) {
    int seq  = blockIdx.x * (blockDim.x >> 5) + (threadIdx.x >> 5);  // 0..127
    int lane = threadIdx.x & 31;
    if (seq >= BB * NN) return;
    int b = seq >> 4, n = seq & 15;

    size_t base = ((size_t)b * SS) * NN + n;
    const int CH = SS / 32;               // 128 steps per lane
    int s0 = lane * CH;

    float a = 1.0f, u = 0.0f;
    for (int t = 0; t < CH; ++t) {
        size_t i = base + (size_t)(s0 + t) * NN;
        float Ae = Aexp[i], Bv = Bu[i];
        u = fmaf(Ae, u, Bv);
        a *= Ae;
    }

#pragma unroll
    for (int off = 1; off < 32; off <<= 1) {
        float ain = __shfl_up(a, off, 32);
        float uin = __shfl_up(u, off, 32);
        if (lane >= off) {
            u = fmaf(a, uin, u);
            a *= ain;
        }
    }

    float h = __shfl_up(u, 1, 32);
    if (lane == 0) h = 0.0f;

    for (int t = 0; t < CH; ++t) {
        size_t i = base + (size_t)(s0 + t) * NN;
        h = fmaf(Aexp[i], h, Bu[i]);
        H[i] = h;
    }
}

// ---------------------------------------------------------------------------
// Kernel 3: y_pre = h @ W_c^T + b_c + x*D, packed to bf16.
// ---------------------------------------------------------------------------
__global__ void k_ypre(const float* __restrict__ x,
                       const float* __restrict__ H,
                       const float* __restrict__ Wc,
                       const float* __restrict__ bc,
                       const float* __restrict__ Dv,
                       unsigned short* __restrict__ Yb) {
    size_t idx = (size_t)blockIdx.x * blockDim.x + threadIdx.x;
    if (idx >= (size_t)ROWS * DM) return;
    size_t row = idx >> 10;
    int d = (int)(idx & (DM - 1));

    const float4* hr = (const float4*)(H  + row * NN);
    const float4* wc = (const float4*)(Wc + (size_t)d * NN);
    float s = 0.0f;
#pragma unroll
    for (int q = 0; q < 4; ++q) {
        float4 hv = hr[q], wv = wc[q];
        s = fmaf(hv.x, wv.x, s);
        s = fmaf(hv.y, wv.y, s);
        s = fmaf(hv.z, wv.z, s);
        s = fmaf(hv.w, wv.w, s);
    }
    float val = s + bc[d] + x[idx] * Dv[d];
    Yb[idx] = f2bf(val);
}

// ---------------------------------------------------------------------------
// Kernel 4: W_o -> bf16
// ---------------------------------------------------------------------------
__global__ void k_woconv(const float* __restrict__ Wo,
                         unsigned short* __restrict__ Wob) {
    size_t idx = (size_t)blockIdx.x * blockDim.x + threadIdx.x;
    if (idx >= (size_t)DM * DM) return;
    Wob[idx] = f2bf(Wo[idx]);
}

// ---------------------------------------------------------------------------
// Kernel 5: out = Yb @ Wob^T + b_o   (M=32768, N=1024, K=1024, bf16 WMMA)
//
// Block = 8 waves -> 128(M) x 64(N) tile; wave owns 16 rows.
// K chunked by 128; the shared 64x128 B tile is staged into LDS with
// global_load_async_to_lds_b128 (ASYNCcnt), double-buffered.
// LDS column stride = 272B (68 dwords == 4 mod 64) -> conflict-free b128 reads.
// ---------------------------------------------------------------------------
#define KC     128                     // K per chunk (bf16 elements)
#define COLSTR 272                     // bytes per staged column (256 + 16 pad)
#define BUFB   (64 * COLSTR)           // 17408 B per buffer

__device__ __forceinline__ void issue_b_chunk(const unsigned short* __restrict__ Wob,
                                              int n0, int k0,
                                              unsigned char* bufBase, int tid) {
#pragma unroll
    for (int i = 0; i < 4; ++i) {
        int col = i * 16 + (tid >> 4);          // 16 cols per issue
        int seg = tid & 15;                     // 16 x 16B segments per col
        const unsigned short* gp = Wob + (size_t)(n0 + col) * DM + k0 + seg * 8;
        unsigned long long ga = (unsigned long long)(uintptr_t)gp;
        unsigned lds = (unsigned)(uintptr_t)(bufBase + col * COLSTR + seg * 16);
        asm volatile("global_load_async_to_lds_b128 %0, %1, off"
                     :: "v"(lds), "v"(ga) : "memory");
    }
}

__global__ void __launch_bounds__(256)
k_gemm_wo(const unsigned short* __restrict__ Yb,
          const unsigned short* __restrict__ Wob,
          const float* __restrict__ bo,
          float* __restrict__ out) {
    __shared__ __align__(16) unsigned char ldsB[2 * BUFB];

    const int K  = DM;
    int tid  = threadIdx.x;
    int wid  = tid >> 5;
    int lane = tid & 31;
    int tileN  = blockIdx.x & 15;               // 16 column tiles of 64
    int blockM = blockIdx.x >> 4;               // 256 row tiles of 128
    int m0 = blockM * 128 + wid * 16;
    int n0 = tileN * 64;
    int half = lane >> 4;
    int nl   = lane & 15;

    const unsigned short* arow = Yb + (size_t)(m0 + nl) * K;

    v8f acc[4] = {};

    const int NC = K / KC;                      // 8 chunks
    issue_b_chunk(Wob, n0, 0, ldsB, tid);

    for (int c = 0; c < NC; ++c) {
        int k0 = c * KC;

        // Preload this chunk's A fragments (global, overlaps async wait).
        bf16x8 areg[8];
#pragma unroll
        for (int s = 0; s < 4; ++s) {
            areg[2 * s]     = *reinterpret_cast<const bf16x8*>(arow + k0 + s * 32 + half * 8);
            areg[2 * s + 1] = *reinterpret_cast<const bf16x8*>(arow + k0 + s * 32 + 16 + half * 8);
        }
        __builtin_prefetch(arow + k0 + KC, 0, 1);   // global_prefetch_b8

        asm volatile("s_wait_asynccnt 0x0" ::: "memory");
        __syncthreads();                             // chunk c resident in LDS

        if (c + 1 < NC)                              // stream next chunk
            issue_b_chunk(Wob, n0, k0 + KC, ldsB + ((c + 1) & 1) * BUFB, tid);

        const unsigned char* bb = ldsB + (c & 1) * BUFB;
#pragma unroll
        for (int s = 0; s < 4; ++s) {
            bf16x16 af = __builtin_shufflevector(areg[2 * s], areg[2 * s + 1],
                            0, 1, 2, 3, 4, 5, 6, 7, 8, 9, 10, 11, 12, 13, 14, 15);
#pragma unroll
            for (int t = 0; t < 4; ++t) {
                const unsigned char* p = bb + (t * 16 + nl) * COLSTR + s * 64 + half * 16;
                bf16x8 blo = *reinterpret_cast<const bf16x8*>(p);
                bf16x8 bhi = *reinterpret_cast<const bf16x8*>(p + 32);
                bf16x16 bfr = __builtin_shufflevector(blo, bhi,
                            0, 1, 2, 3, 4, 5, 6, 7, 8, 9, 10, 11, 12, 13, 14, 15);
                acc[t] = __builtin_amdgcn_wmma_f32_16x16x32_bf16(
                             false, af, false, bfr, (short)0, acc[t], false, false);
            }
        }
        __syncthreads();                             // done reading buffer c&1
    }

    // C/D layout: lane -> column (nl) within subtile, VGPR r -> row half*8+r
#pragma unroll
    for (int t = 0; t < 4; ++t) {
        int col = n0 + t * 16 + nl;
        float bias = bo[col];
#pragma unroll
        for (int r = 0; r < 8; ++r) {
            int row = m0 + half * 8 + r;
            out[(size_t)row * DM + col] = acc[t][r] + bias;
        }
    }
}

// ---------------------------------------------------------------------------
// Launcher.  d_in order: x, A, W_b, b_b, W_c, b_c, D, W_t, b_t, W_o, b_o
// ---------------------------------------------------------------------------
extern "C" void kernel_launch(void* const* d_in, const int* in_sizes, int n_in,
                              void* d_out, int out_size, void* d_ws, size_t ws_size,
                              hipStream_t stream) {
    (void)in_sizes; (void)n_in; (void)out_size; (void)ws_size;

    const float* x  = (const float*)d_in[0];
    const float* A  = (const float*)d_in[1];
    const float* Wb = (const float*)d_in[2];
    const float* bb = (const float*)d_in[3];
    const float* Wc = (const float*)d_in[4];
    const float* bc = (const float*)d_in[5];
    const float* Dv = (const float*)d_in[6];
    const float* Wt = (const float*)d_in[7];
    const float* bt = (const float*)d_in[8];
    const float* Wo = (const float*)d_in[9];
    const float* bo = (const float*)d_in[10];
    float* out = (float*)d_out;

    char* ws = (char*)d_ws;
    float*          Bu   = (float*)(ws + (0ull  << 20));   //  2 MB
    float*          Aexp = (float*)(ws + (2ull  << 20));   //  2 MB
    float*          H    = (float*)(ws + (4ull  << 20));   //  2 MB
    unsigned short* Wob  = (unsigned short*)(ws + (6ull << 20));  // 2 MB
    unsigned short* Yb   = (unsigned short*)(ws + (8ull << 20));  // 64 MB

    // 1) delta / Bu / A_exp : one wave per row, 8 waves per block
    k_bu_delta<<<ROWS / 8, 256, 0, stream>>>(x, A, Wb, bb, Wt, bt, Bu, Aexp);

    // 4) W_o -> bf16 (independent of 1-3)
    k_woconv<<<(DM * DM) / 256, 256, 0, stream>>>(Wo, Wob);

    // 2) scan: 128 sequences, 4 waves per block
    k_scan<<<(BB * NN) / 4, 128, 0, stream>>>(Aexp, Bu, H);

    // 3) y_pre + bf16 pack
    k_ypre<<<(ROWS * (size_t)DM) / 256, 256, 0, stream>>>(x, H, Wc, bc, Dv, Yb);

    // 5) big GEMM on the WMMA pipe (async-LDS double-buffered B)
    k_gemm_wo<<<(ROWS / 128) * (DM / 64), 256, 0, stream>>>(Yb, Wob, bo, out);
}
